// NodeReconstructionGINE_19808389169618
// MI455X (gfx1250) — matrix operand order repaired and checked by reading, not compiled, not run on hardware
//
#include <hip/hip_runtime.h>
#include <hip/hip_bf16.h>

// ---------------------------------------------------------------------------
// GINE forward for MI455X (gfx1250, wave32, WMMA 16x16x32 bf16).
// - Edge path fused: bond-encoder GEMMs + message + scatter-add, one kernel.
// - Weights pre-swizzled into WMMA B-fragment order in LDS -> ds_load_b128.
// - A operands loaded as float4 / v8bf vectors.
// ---------------------------------------------------------------------------

typedef __attribute__((ext_vector_type(16))) __bf16 v16bf;
typedef __attribute__((ext_vector_type(8)))  __bf16 v8bf;
typedef __attribute__((ext_vector_type(8)))  float  v8f;

#define N_NODES 100000
#define N_EDGES 1000000
#define N_GRAPH 1024
#define N_SUB   128
#define HDIM    64
#define EDIM    32
#define ODIM    32
#define NLAYER  3

__device__ __forceinline__ v8f wmma_bf16_acc(v16bf a, v16bf b, v8f c) {
    // D = A(16x32 bf16) * B(32x16 bf16) + C(16x16 f32)
    return __builtin_amdgcn_wmma_f32_16x16x32_bf16(
        false, a, false, b, (short)0, c, false, false);
}

// Fragment lane mappings (wave32):
//  A (16x32 MxK): lane L row M=L&15; elem e: K=(e&7)+(e&8?16:0)+8*(L>>4)
//                 -> elems 0..7 / 8..15 are two contiguous 8-runs in K.
//  B (32x16 KxN): lane L col N=L&15; elem e: K=(e&7)+(e&8?16:0)+8*(L>>4)
//  C (16x16 f32): lane L col N=L&15; VGPR r: row M=r+8*(L>>4)

// Pre-swizzled B-fragment LDS layout: frag[((kb*4+t)*32+lane)*16 + e]
__device__ __forceinline__ void fill_bfrag(__bf16* frag, const float* w,
                                           int nelem, int tid) {
    // nelem = kbs*4*32*16 ; w is [K,64] row-major
    for (int i = tid; i < nelem; i += 256) {
        int e  = i & 15;
        int ln = (i >> 4) & 31;
        int t  = (i >> 9) & 3;
        int kb = i >> 11;
        int K   = (e & 7) + ((e & 8) ? 16 : 0) + ((ln >> 4) << 3) + kb * 32;
        int col = t * 16 + (ln & 15);
        frag[i] = (__bf16)w[K * HDIM + col];
    }
}

__device__ __forceinline__ v16bf load_bfrag(const __bf16* frag, int kb, int t,
                                            int lane) {
    return *(const v16bf*)(frag + (((kb << 2) | t) * 32 + lane) * 16);
}

__device__ __forceinline__ float4 ld4(const float* p) {
    return *(const float4*)p;
}

__device__ __forceinline__ v16bf pack_a_f32(float4 q0, float4 q1, float4 q2,
                                            float4 q3) {
    v16bf a;
    a[0]  = (__bf16)q0.x; a[1]  = (__bf16)q0.y; a[2]  = (__bf16)q0.z; a[3]  = (__bf16)q0.w;
    a[4]  = (__bf16)q1.x; a[5]  = (__bf16)q1.y; a[6]  = (__bf16)q1.z; a[7]  = (__bf16)q1.w;
    a[8]  = (__bf16)q2.x; a[9]  = (__bf16)q2.y; a[10] = (__bf16)q2.z; a[11] = (__bf16)q2.w;
    a[12] = (__bf16)q3.x; a[13] = (__bf16)q3.y; a[14] = (__bf16)q3.z; a[15] = (__bf16)q3.w;
    return a;
}

// ---------------------------------------------------------------------------
// Kernel 1 (per layer): fused bond encoder + message + scatter-add.
// 8 waves/block, 16 edges/wave -> 128 edges/block.
// ---------------------------------------------------------------------------
__global__ __launch_bounds__(256) void gine_edge_kernel(
    const float* __restrict__ hprev,       // [N,64]
    const float* __restrict__ edge_attr,   // [E,32]
    const int*   __restrict__ src,
    const int*   __restrict__ dst,
    const float* __restrict__ w1,          // [32,64]
    const float* __restrict__ b1,          // [64]
    const float* __restrict__ w2,          // [64,64]
    const float* __restrict__ b2,          // [64]
    float*       __restrict__ agg)         // [N,64]
{
    __shared__ __attribute__((aligned(32))) __bf16 fw1[1 * 4 * 32 * 16]; // K=32
    __shared__ __attribute__((aligned(32))) __bf16 fw2[2 * 4 * 32 * 16]; // K=64
    __shared__ __attribute__((aligned(16))) __bf16 tile[8][16][HDIM + 8];

    const int tid = threadIdx.x;
    fill_bfrag(fw1, w1, 2048, tid);
    fill_bfrag(fw2, w2, 4096, tid);
    __syncthreads();

    const int lane  = tid & 31;
    const int wid   = tid >> 5;
    const long long e0 = (long long)blockIdx.x * 128 + (long long)wid * 16;

    const int mrow  = lane & 15;
    const int khalf = (lane >> 4) * 8;
    const int ncol  = lane & 15;
    const int rbase = (lane >> 4) * 8;

    // ---- stage 1: A = edge_attr tile (16 x 32), vector loads ----
    long long erow = e0 + mrow;
    erow = (erow < N_EDGES) ? erow : (long long)(N_EDGES - 1);
    const float* ea = edge_attr + erow * EDIM;
    v16bf a1 = pack_a_f32(ld4(ea + khalf),      ld4(ea + khalf + 4),
                          ld4(ea + khalf + 16), ld4(ea + khalf + 20));

    v8f acc[4];
#pragma unroll
    for (int t = 0; t < 4; ++t) {
        v8f z = {};
        acc[t] = wmma_bf16_acc(a1, load_bfrag(fw1, 0, t, lane), z);
    }

    // bias + relu -> LDS (relayout C -> A for next GEMM)
#pragma unroll
    for (int t = 0; t < 4; ++t) {
        int col = t * 16 + ncol;
        float bias = b1[col];
#pragma unroll
        for (int r = 0; r < 8; ++r) {
            float v = acc[t][r] + bias;
            tile[wid][rbase + r][col] = (__bf16)fmaxf(v, 0.0f);
        }
    }
    __syncthreads();

    // ---- stage 2: tile(16x64) @ W2(64x64), two 32-K chunks ----
    v8f acc2[4];
#pragma unroll
    for (int t = 0; t < 4; ++t) { v8f z = {}; acc2[t] = z; }
#pragma unroll
    for (int kb = 0; kb < 2; ++kb) {
        v8bf tlo = *(const v8bf*)&tile[wid][mrow][khalf + kb * 32];
        v8bf thi = *(const v8bf*)&tile[wid][mrow][khalf + 16 + kb * 32];
        v16bf a2;
#pragma unroll
        for (int e = 0; e < 8; ++e) { a2[e] = tlo[e]; a2[e + 8] = thi[e]; }
#pragma unroll
        for (int t = 0; t < 4; ++t)
            acc2[t] = wmma_bf16_acc(a2, load_bfrag(fw2, kb, t, lane), acc2[t]);
    }

    // ---- stage 3: msg = relu(h[src] + e) ; scatter-add to agg[dst] ----
    int  sidx[8], didx[8];
    bool valid[8];
#pragma unroll
    for (int r = 0; r < 8; ++r) {
        long long e = e0 + rbase + r;
        valid[r] = (e < N_EDGES);
        long long ec = valid[r] ? e : (long long)(N_EDGES - 1);
        sidx[r] = src[ec];
        didx[r] = dst[ec];
    }
#pragma unroll
    for (int t = 0; t < 4; ++t) {
        int col = t * 16 + ncol;
        float bias = b2[col];
#pragma unroll
        for (int r = 0; r < 8; ++r) {
            if (valid[r]) {
                float hv  = hprev[(long long)sidx[r] * HDIM + col];
                float msg = fmaxf(hv + acc2[t][r] + bias, 0.0f);
                atomicAdd(&agg[(long long)didx[r] * HDIM + col], msg);
            }
        }
    }
}

// ---------------------------------------------------------------------------
// Kernel 2 (per layer): node update MLP (two WMMA GEMMs) + BN statistics.
// ---------------------------------------------------------------------------
__global__ __launch_bounds__(256) void gine_node_kernel(
    const float* __restrict__ hprev,
    const float* __restrict__ agg,
    const float* __restrict__ w1, const float* __restrict__ b1,
    const float* __restrict__ w2, const float* __restrict__ b2,
    const float* __restrict__ eps_p, int layer,
    float* __restrict__ zbuf, float* __restrict__ stats)
{
    __shared__ __attribute__((aligned(32))) __bf16 fw1[2 * 4 * 32 * 16];
    __shared__ __attribute__((aligned(32))) __bf16 fw2[2 * 4 * 32 * 16];
    __shared__ __attribute__((aligned(16))) __bf16 tile[8][16][HDIM + 8];

    const int tid = threadIdx.x;
    fill_bfrag(fw1, w1, 4096, tid);
    fill_bfrag(fw2, w2, 4096, tid);
    __syncthreads();

    const int lane  = tid & 31;
    const int wid   = tid >> 5;
    const long long n0 = (long long)blockIdx.x * 128 + (long long)wid * 16;

    const int mrow  = lane & 15;
    const int khalf = (lane >> 4) * 8;
    const int ncol  = lane & 15;
    const int rbase = (lane >> 4) * 8;

    const float ep = 1.0f + eps_p[layer];

    long long nrow = n0 + mrow;
    nrow = (nrow < N_NODES) ? nrow : (long long)(N_NODES - 1);
    const float* hr = hprev + nrow * HDIM;
    const float* ar = agg   + nrow * HDIM;

    // ---- GEMM 1: ((1+eps)h + agg) @ W1, vector loads ----
    v8f acc[4];
#pragma unroll
    for (int t = 0; t < 4; ++t) { v8f z = {}; acc[t] = z; }
#pragma unroll
    for (int kb = 0; kb < 2; ++kb) {
        const int k0 = khalf + kb * 32;
        float4 h0 = ld4(hr + k0),      h1 = ld4(hr + k0 + 4);
        float4 h2 = ld4(hr + k0 + 16), h3 = ld4(hr + k0 + 20);
        float4 g0 = ld4(ar + k0),      g1 = ld4(ar + k0 + 4);
        float4 g2 = ld4(ar + k0 + 16), g3 = ld4(ar + k0 + 20);
        float4 q0 = make_float4(ep*h0.x+g0.x, ep*h0.y+g0.y, ep*h0.z+g0.z, ep*h0.w+g0.w);
        float4 q1 = make_float4(ep*h1.x+g1.x, ep*h1.y+g1.y, ep*h1.z+g1.z, ep*h1.w+g1.w);
        float4 q2 = make_float4(ep*h2.x+g2.x, ep*h2.y+g2.y, ep*h2.z+g2.z, ep*h2.w+g2.w);
        float4 q3 = make_float4(ep*h3.x+g3.x, ep*h3.y+g3.y, ep*h3.z+g3.z, ep*h3.w+g3.w);
        v16bf a = pack_a_f32(q0, q1, q2, q3);
#pragma unroll
        for (int t = 0; t < 4; ++t)
            acc[t] = wmma_bf16_acc(a, load_bfrag(fw1, kb, t, lane), acc[t]);
    }

    // bias + relu -> LDS relayout
#pragma unroll
    for (int t = 0; t < 4; ++t) {
        int col = t * 16 + ncol;
        float bias = b1[col];
#pragma unroll
        for (int r = 0; r < 8; ++r) {
            float v = acc[t][r] + bias;
            tile[wid][rbase + r][col] = (__bf16)fmaxf(v, 0.0f);
        }
    }
    __syncthreads();

    // ---- GEMM 2 ----
    v8f acc2[4];
#pragma unroll
    for (int t = 0; t < 4; ++t) { v8f z = {}; acc2[t] = z; }
#pragma unroll
    for (int kb = 0; kb < 2; ++kb) {
        v8bf tlo = *(const v8bf*)&tile[wid][mrow][khalf + kb * 32];
        v8bf thi = *(const v8bf*)&tile[wid][mrow][khalf + 16 + kb * 32];
        v16bf a2;
#pragma unroll
        for (int e = 0; e < 8; ++e) { a2[e] = tlo[e]; a2[e + 8] = thi[e]; }
#pragma unroll
        for (int t = 0; t < 4; ++t)
            acc2[t] = wmma_bf16_acc(a2, load_bfrag(fw2, kb, t, lane), acc2[t]);
    }

    // ---- bias + relu, store pre-BN z, accumulate BN statistics ----
#pragma unroll
    for (int t = 0; t < 4; ++t) {
        int col = t * 16 + ncol;
        float bias = b2[col];
        float s = 0.0f, s2 = 0.0f;
#pragma unroll
        for (int r = 0; r < 8; ++r) {
            long long node = n0 + rbase + r;
            float v = fmaxf(acc2[t][r] + bias, 0.0f);
            if (node < N_NODES) {
                zbuf[node * HDIM + col] = v;
                s  += v;
                s2 += v * v;
            }
        }
        atomicAdd(&stats[col], s);
        atomicAdd(&stats[HDIM + col], s2);
    }
}

// ---------------------------------------------------------------------------
// Kernel 3 (per layer): apply BN, fused graph mean-pool scatter. 4 cols/thread.
// ---------------------------------------------------------------------------
__global__ __launch_bounds__(256) void gine_bn_pool_kernel(
    const float* __restrict__ zbuf, const float* __restrict__ stats,
    const float* __restrict__ gamma, const float* __restrict__ beta,
    const int*   __restrict__ batch,
    float* __restrict__ hbuf, float* __restrict__ gsum, float* __restrict__ cnt,
    int layer)
{
    long long idx = (long long)blockIdx.x * blockDim.x + threadIdx.x;
    if (idx >= (long long)N_NODES * (HDIM / 4)) return;
    int cq = (int)(idx & (HDIM / 4 - 1));    // column quad 0..15
    long long node = idx >> 4;
    int c = cq * 4;

    const float invN = 1.0f / (float)N_NODES;
    float4 z = ld4(zbuf + node * HDIM + c);
    float4 o;
    {
        float mu  = stats[c + 0] * invN;
        float var = stats[HDIM + c + 0] * invN - mu * mu;
        o.x = (z.x - mu) * rsqrtf(var + 1e-5f) * gamma[c + 0] + beta[c + 0];
        mu  = stats[c + 1] * invN;
        var = stats[HDIM + c + 1] * invN - mu * mu;
        o.y = (z.y - mu) * rsqrtf(var + 1e-5f) * gamma[c + 1] + beta[c + 1];
        mu  = stats[c + 2] * invN;
        var = stats[HDIM + c + 2] * invN - mu * mu;
        o.z = (z.z - mu) * rsqrtf(var + 1e-5f) * gamma[c + 2] + beta[c + 2];
        mu  = stats[c + 3] * invN;
        var = stats[HDIM + c + 3] * invN - mu * mu;
        o.w = (z.w - mu) * rsqrtf(var + 1e-5f) * gamma[c + 3] + beta[c + 3];
    }
    *(float4*)(hbuf + node * HDIM + c) = o;

    int g = batch[node];
    float* gp = gsum + (long long)g * (NLAYER * HDIM) + layer * HDIM + c;
    atomicAdd(gp + 0, o.x);
    atomicAdd(gp + 1, o.y);
    atomicAdd(gp + 2, o.z);
    atomicAdd(gp + 3, o.w);
    if (layer == 0 && cq == 0) atomicAdd(&cnt[g], 1.0f);
}

// ---------------------------------------------------------------------------
// Kernel 4: per-graph fc0 + relu * weight, scatter into subgraph sums.
// ---------------------------------------------------------------------------
__global__ __launch_bounds__(64) void gine_graph_kernel(
    const float* __restrict__ gsum, const float* __restrict__ cnt,
    const float* __restrict__ fc0_w, const float* __restrict__ fc0_b,
    const float* __restrict__ wts, const int* __restrict__ sub_batch,
    float* __restrict__ ssum, float* __restrict__ norm)
{
    __shared__ float grow[NLAYER * HDIM];
    int g = blockIdx.x;
    int c = threadIdx.x;
    float cv = fmaxf(cnt[g], 1.0f);
    for (int i = c; i < NLAYER * HDIM; i += 64)
        grow[i] = gsum[(long long)g * (NLAYER * HDIM) + i] / cv;
    __syncthreads();

    float acc = fc0_b[c];
    for (int k = 0; k < NLAYER * HDIM; ++k)
        acc += grow[k] * fc0_w[k * HDIM + c];
    float w = wts[g];
    acc = fmaxf(acc, 0.0f) * w;
    int sb = sub_batch[g];
    atomicAdd(&ssum[sb * HDIM + c], acc);
    if (c == 0) atomicAdd(&norm[sb], w);
}

// ---------------------------------------------------------------------------
// Kernel 5: subgraph normalize + fc1 + fc2 + prediction head.
// ---------------------------------------------------------------------------
__global__ __launch_bounds__(64) void gine_final_kernel(
    const float* __restrict__ ssum, const float* __restrict__ norm,
    const float* __restrict__ fc1_w, const float* __restrict__ fc1_b,
    const float* __restrict__ fc2_w, const float* __restrict__ fc2_b,
    const float* __restrict__ pw, const float* __restrict__ pb,
    float* __restrict__ out)
{
    __shared__ float sA[HDIM], sB[HDIM];
    int sgi = blockIdx.x;
    int c = threadIdx.x;

    float nv = fmaxf(norm[sgi], 1e-12f);
    sA[c] = ssum[sgi * HDIM + c] / nv;
    __syncthreads();

    float a = fc1_b[c];
    for (int k = 0; k < HDIM; ++k) a += sA[k] * fc1_w[k * HDIM + c];
    sB[c] = fmaxf(a, 0.0f);
    __syncthreads();

    a = fc2_b[c];
    for (int k = 0; k < HDIM; ++k) a += sB[k] * fc2_w[k * HDIM + c];
    __syncthreads();
    sA[c] = fmaxf(a, 0.0f);
    __syncthreads();

    if (c < ODIM) {
        float o = pb[c];
        for (int k = 0; k < HDIM; ++k) o += sA[k] * pw[k * ODIM + c];
        out[sgi * ODIM + c] = o;
    }
}

// ---------------------------------------------------------------------------
// Launch
// ---------------------------------------------------------------------------
extern "C" void kernel_launch(void* const* d_in, const int* in_sizes, int n_in,
                              void* d_out, int out_size, void* d_ws, size_t ws_size,
                              hipStream_t stream) {
    const float* x         = (const float*)d_in[0];
    const int*   eidx      = (const int*)  d_in[1];
    const float* edge_attr = (const float*)d_in[2];
    const int*   batch     = (const int*)  d_in[3];
    const float* wts       = (const float*)d_in[4];
    const int*   sub_b     = (const int*)  d_in[5];
    const float* be_w1     = (const float*)d_in[6];
    const float* be_b1     = (const float*)d_in[7];
    const float* be_w2     = (const float*)d_in[8];
    const float* be_b2     = (const float*)d_in[9];
    const float* mlp_w1    = (const float*)d_in[10];
    const float* mlp_b1    = (const float*)d_in[11];
    const float* mlp_w2    = (const float*)d_in[12];
    const float* mlp_b2    = (const float*)d_in[13];
    const float* eps       = (const float*)d_in[14];
    const float* gamma     = (const float*)d_in[15];
    const float* beta      = (const float*)d_in[16];
    const float* fc0_w     = (const float*)d_in[17];
    const float* fc0_b     = (const float*)d_in[18];
    const float* fc1_w     = (const float*)d_in[19];
    const float* fc1_b     = (const float*)d_in[20];
    const float* fc2_w     = (const float*)d_in[21];
    const float* fc2_b     = (const float*)d_in[22];
    const float* pred_w    = (const float*)d_in[23];
    const float* pred_b    = (const float*)d_in[24];

    const int* src = eidx;
    const int* dst = eidx + N_EDGES;

    char* ws = (char*)d_ws;
    size_t off = 0;
    auto carve = [&](size_t bytes) {
        void* p = ws + off;
        off += (bytes + 255) & ~(size_t)255;
        return p;
    };
    float* agg   = (float*)carve((size_t)N_NODES * HDIM * sizeof(float));
    float* zbuf  = (float*)carve((size_t)N_NODES * HDIM * sizeof(float));
    float* hbuf  = (float*)carve((size_t)N_NODES * HDIM * sizeof(float));
    float* gsum  = (float*)carve((size_t)N_GRAPH * NLAYER * HDIM * sizeof(float));
    float* cnt   = (float*)carve((size_t)N_GRAPH * sizeof(float));
    float* ssum  = (float*)carve((size_t)N_SUB * HDIM * sizeof(float));
    float* norm  = (float*)carve((size_t)N_SUB * sizeof(float));
    float* stats = (float*)carve((size_t)2 * HDIM * sizeof(float));

    hipMemsetAsync(gsum, 0, (size_t)N_GRAPH * NLAYER * HDIM * sizeof(float), stream);
    hipMemsetAsync(cnt,  0, (size_t)N_GRAPH * sizeof(float), stream);
    hipMemsetAsync(ssum, 0, (size_t)N_SUB * HDIM * sizeof(float), stream);
    hipMemsetAsync(norm, 0, (size_t)N_SUB * sizeof(float), stream);

    dim3 blk(256);
    const int egrid = (N_EDGES + 127) / 128;
    const int ngrid = (N_NODES + 127) / 128;
    const int bgrid = (int)(((long long)N_NODES * (HDIM / 4) + 255) / 256);

    const float* hprev = x;
    for (int l = 0; l < NLAYER; ++l) {
        hipMemsetAsync(agg,   0, (size_t)N_NODES * HDIM * sizeof(float), stream);
        hipMemsetAsync(stats, 0, (size_t)2 * HDIM * sizeof(float), stream);

        gine_edge_kernel<<<egrid, blk, 0, stream>>>(
            hprev, edge_attr, src, dst,
            be_w1 + (size_t)l * EDIM * HDIM, be_b1 + (size_t)l * HDIM,
            be_w2 + (size_t)l * HDIM * HDIM, be_b2 + (size_t)l * HDIM,
            agg);

        gine_node_kernel<<<ngrid, blk, 0, stream>>>(
            hprev, agg,
            mlp_w1 + (size_t)l * HDIM * HDIM, mlp_b1 + (size_t)l * HDIM,
            mlp_w2 + (size_t)l * HDIM * HDIM, mlp_b2 + (size_t)l * HDIM,
            eps, l, zbuf, stats);

        gine_bn_pool_kernel<<<bgrid, blk, 0, stream>>>(
            zbuf, stats, gamma + (size_t)l * HDIM, beta + (size_t)l * HDIM,
            batch, hbuf, gsum, cnt, l);

        hprev = hbuf;
    }

    gine_graph_kernel<<<N_GRAPH, 64, 0, stream>>>(
        gsum, cnt, fc0_w, fc0_b, wts, sub_b, ssum, norm);

    gine_final_kernel<<<N_SUB, 64, 0, stream>>>(
        ssum, norm, fc1_w, fc1_b, fc2_w, fc2_b, pred_w, pred_b, (float*)d_out);
}
